// Grouper4_25039659335966
// MI455X (gfx1250) — matrix operand order repaired and checked by reading, not compiled, not run on hardware
//
#include <hip/hip_runtime.h>
#include <hip/hip_bf16.h>

typedef __attribute__((ext_vector_type(16))) _Float16 v16h;
typedef __attribute__((ext_vector_type(8)))  _Float16 v8h;
typedef __attribute__((ext_vector_type(8)))  float    v8f;
typedef int v4i_t __attribute__((vector_size(16)));

#define N_PTS   16384
#define M_QRY   4096
#define B_BATCH 2
#define KSAMP   32
#define CIN     64
#define ROWS    (B_BATCH * M_QRY * KSAMP)   // 262144
#define R2      0.04f

#define AS1 __attribute__((address_space(1)))
#define AS3 __attribute__((address_space(3)))

#if defined(__has_builtin)
#if __has_builtin(__builtin_amdgcn_global_load_async_to_lds_b128) && \
    __has_builtin(__builtin_amdgcn_s_wait_asynccnt)
#define HAVE_ASYNC_LDS 1
#endif
#endif

// async (or fallback) block copy of n16chunks 16B chunks from a global f16
// panel -> LDS, distributed over 256 threads. Async path is ASYNCcnt-tracked
// (global_load_async_to_lds_b128) and is drained with s_wait_asynccnt(0)
// before returning, so it does not overlap later work.
__device__ inline void stage_panel(const _Float16* __restrict__ gsrc,
                                   _Float16* lds, int n16chunks, int t) {
#ifdef HAVE_ASYNC_LDS
    AS1 v4i_t* s = (AS1 v4i_t*)(const void*)gsrc;
    AS3 v4i_t* d = (AS3 v4i_t*)(void*)lds;
    for (int i = t; i < n16chunks; i += 256) {
        __builtin_amdgcn_global_load_async_to_lds_b128(s + i, d + i, 0, 0);
    }
    __builtin_amdgcn_s_wait_asynccnt(0);
#else
    const v8h* s = (const v8h*)gsrc;
    v8h* d = (v8h*)lds;
    for (int i = t; i < n16chunks; i += 256) d[i] = s[i];
#endif
}

// ---------------------------------------------------------------------------
// stats float layout: [0:64) sum1 | [64:128) sq1 | [128:256) sum2 | [256:384) sq2
//                     [384:448) scale1 | [448:512) shift1 | [512:640) scale2 | [640:768) shift2
// ---------------------------------------------------------------------------
__global__ void zero_stats_kernel(float* stats) {
    int t = threadIdx.x;
    if (t < 384) stats[t] = 0.0f;
}

// One-shot: transpose + f16-convert both weight matrices into ws panels.
// Wt1[o][k] (64x96, K zero-padded past 67), Wt2[o][k] (128x64).
__global__ void prep_weights_kernel(const float* __restrict__ W1,
                                    const float* __restrict__ W2,
                                    _Float16* __restrict__ wt1,
                                    _Float16* __restrict__ wt2) {
    const int t = threadIdx.x;
    for (int i = t; i < 64 * 96; i += 256) {
        const int o = i / 96, k = i % 96;
        wt1[o * 96 + k] = (_Float16)((k < 67) ? W1[k * 64 + o] : 0.0f);
    }
    for (int i = t; i < 128 * 64; i += 256) {
        const int o = i >> 6, k = i & 63;
        wt2[o * 64 + k] = (_Float16)W2[k * 128 + o];
    }
}

// ---------------------------------------------------------------------------
// Kernel 1: ball query. One wave32 per query; ordered compaction via
// ballot/popc; early exit once 32 neighbors found. Emits idx, rel(xyz), rec.
// ---------------------------------------------------------------------------
__global__ __launch_bounds__(256) void ballquery_kernel(
    const float* __restrict__ xyz, const float* __restrict__ newxyz,
    int* __restrict__ idxW, float* __restrict__ recW, float* __restrict__ relW)
{
    __shared__ int sidx[8][KSAMP];
    const int t = threadIdx.x;
    const int wv = t >> 5, lane = t & 31;
    const long qid = (long)blockIdx.x * 8 + wv;       // 0..8191
    const int b = (int)(qid >> 12);
    const float* xb = xyz + (long)b * N_PTS * 3;

    const float qx = newxyz[qid * 3 + 0];
    const float qy = newxyz[qid * 3 + 1];
    const float qz = newxyz[qid * 3 + 2];
    const float qq = qx * qx + qy * qy + qz * qz;

    int cnt = 0;
    for (int base = 0; base < N_PTS && cnt < KSAMP; base += 32) {
        const int i = base + lane;
        const float px = xb[i * 3 + 0];
        const float py = xb[i * 3 + 1];
        const float pz = xb[i * 3 + 2];
        const float pp = px * px + py * py + pz * pz;
        const float d2 = qq + pp - 2.0f * (qx * px + qy * py + qz * pz);
        const bool hit = d2 < R2;
        const unsigned bal = __builtin_amdgcn_ballot_w32(hit);
        if (hit) {
            const int pos = cnt + __popc(bal & ((1u << lane) - 1u));
            if (pos < KSAMP) sidx[wv][pos] = i;
        }
        cnt += __popc(bal);
    }
    if (cnt > KSAMP) cnt = KSAMP;
    __syncthreads();

    const int id = (lane < cnt) ? sidx[wv][lane] : (cnt > 0 ? sidx[wv][0] : 0);
    const float px = xb[id * 3 + 0], py = xb[id * 3 + 1], pz = xb[id * 3 + 2];
    const float rx = px - qx, ry = py - qy, rz = pz - qz;
    const float d = sqrtf(rx * rx + ry * ry + rz * rz);

    const long row = qid * KSAMP + lane;
    idxW[row] = id;
    recW[row] = 1.0f / (d + 1e-8f);
    relW[row * 3 + 0] = rx;
    relW[row * 3 + 1] = ry;
    relW[row * 3 + 2] = rz;
}

// ---------------------------------------------------------------------------
// WMMA fragment loaders (wave32 layouts per CDNA5 ISA 7.12.2).
// A 16x32 f16: lane L -> row = L&15; K = (L>>4)*8 + {0..7} and 16+(L>>4)*8+{0..7}
// B 32x16 f16 (stored transposed Wt[col][k]): lane L -> col = L&15;
//   K = (L>>4)*8 + {0..7} and 16+(L>>4)*8+{0..7} (same as A).
// ---------------------------------------------------------------------------
__device__ inline v16h load_a_frag(const _Float16* p /* &A[row][kb] */) {
    v8h lo = *(const v8h*)p;
    v8h hi = *(const v8h*)(p + 16);
    v16h a;
    for (int i = 0; i < 8; ++i) { a[i] = lo[i]; a[i + 8] = hi[i]; }
    return a;
}
__device__ inline v16h load_b_frag(const _Float16* p /* &Wt[col][kb] */) {
    v8h lo = *(const v8h*)p;
    v8h hi = *(const v8h*)(p + 16);
    v16h bfr;
    for (int i = 0; i < 8; ++i) { bfr[i] = lo[i]; bfr[i + 8] = hi[i]; }
    return bfr;
}

// ---------------------------------------------------------------------------
// Kernel 2: layer-1 GEMM. 128 rows (4 queries) per block, 8 waves.
// A = [rel(3) | gathered feat(64) | pad -> 96] in f16 LDS; Wt1 async-staged.
// Wave w computes 16-row strip x 64 cols: 4 col-tiles x 3 k-steps = 12 WMMAs.
// Emits pre-BN z1 (f16) + per-channel sum/sumsq via LDS->global atomics.
// ---------------------------------------------------------------------------
__global__ __launch_bounds__(256) void layer1_kernel(
    const float* __restrict__ feat, const _Float16* __restrict__ wt1g,
    const float* __restrict__ b1, const int* __restrict__ idxW,
    const float* __restrict__ relW, _Float16* __restrict__ z1,
    float* __restrict__ sum1, float* __restrict__ sq1)
{
    __shared__ _Float16 As[128][96];
    __shared__ _Float16 Wt[64][96];
    __shared__ float lsum[64], lsq[64];

    const int t = threadIdx.x;
    const long rowBase = (long)blockIdx.x * 128;

    // stage weight panel; completes (s_wait_asynccnt) before the gathers below
    stage_panel(wt1g, &Wt[0][0], (64 * 96) / 8, t);

    if (t < 64) { lsum[t] = 0.0f; lsq[t] = 0.0f; }

    // Build A: 2 threads per row, 48 channels each (L2-resident gathers)
    {
        const int r = t >> 1;
        const int half = t & 1;
        const long row = rowBase + r;
        const int id = idxW[row];
        const int b = (int)(row >> 17);                 // row / 131072
        const float* fb = feat + (long)b * CIN * N_PTS;
        for (int cc = 0; cc < 48; ++cc) {
            const int ch = half * 48 + cc;
            float v;
            if (ch < 3)        v = relW[row * 3 + ch];
            else if (ch < 67)  v = fb[(long)(ch - 3) * N_PTS + id];
            else               v = 0.0f;
            As[r][ch] = (_Float16)v;
        }
    }
    __syncthreads();

    const int wv = t >> 5, lane = t & 31;
    const int row0 = wv * 16;
    const int cl = lane & 15;
    const int hs = lane >> 4;

    for (int ct = 0; ct < 4; ++ct) {
        v8f acc = {};
        for (int ks = 0; ks < 3; ++ks) {
            const int k0 = ks * 32;
            v16h a = load_a_frag(&As[row0 + cl][k0 + (hs << 3)]);
            v16h bf = load_b_frag(&Wt[ct * 16 + cl][k0 + (hs << 3)]);
            acc = __builtin_amdgcn_wmma_f32_16x16x32_f16(
                false, a, false, bf, (short)0, acc, false, false);
        }
        const int o = ct * 16 + cl;
        const float bias = b1[o];
        float s = 0.0f, s2 = 0.0f;
        for (int r8 = 0; r8 < 8; ++r8) {
            const long r = rowBase + row0 + hs * 8 + r8;
            const float z = acc[r8] + bias;
            z1[r * 64 + o] = (_Float16)z;
            s += z; s2 += z * z;
        }
        atomicAdd(&lsum[o], s);
        atomicAdd(&lsq[o], s2);
    }
    __syncthreads();
    if (t < 64) { atomicAdd(&sum1[t], lsum[t]); atomicAdd(&sq1[t], lsq[t]); }
}

// ---------------------------------------------------------------------------
// Kernel 3: fold BN stats into scale/shift (generic for both layers)
// ---------------------------------------------------------------------------
__global__ void finalize_bn_kernel(
    const float* __restrict__ sum, const float* __restrict__ sq,
    const float* __restrict__ g, const float* __restrict__ be,
    float* __restrict__ scale, float* __restrict__ shift,
    int C, float invCount)
{
    const int c = threadIdx.x;
    if (c < C) {
        const float mu = sum[c] * invCount;
        const float var = sq[c] * invCount - mu * mu;
        const float rs = rsqrtf(var + 1e-5f);
        const float sc = rs * g[c];
        scale[c] = sc;
        shift[c] = be[c] - mu * sc;
    }
}

// ---------------------------------------------------------------------------
// Kernel 4: layer-2 GEMM. 64 rows per block. BN1+ReLU applied while staging
// z1 into LDS as f16; Wt2 panel async-staged. Wave w owns col-tile w,
// iterates 4 row-tiles x 2 k-steps = 8 WMMAs. Emits pre-BN z2 (f16) + stats.
// ---------------------------------------------------------------------------
__global__ __launch_bounds__(256) void layer2_kernel(
    const _Float16* __restrict__ z1, const _Float16* __restrict__ wt2g,
    const float* __restrict__ b2, const float* __restrict__ scale1,
    const float* __restrict__ shift1, _Float16* __restrict__ z2,
    float* __restrict__ sum2, float* __restrict__ sq2)
{
    __shared__ _Float16 As[64][64];
    __shared__ _Float16 Wt[128][64];
    __shared__ float lsum[128], lsq[128];

    const int t = threadIdx.x;
    const long rowBase = (long)blockIdx.x * 64;

    stage_panel(wt2g, &Wt[0][0], (128 * 64) / 8, t);

    if (t < 128) { lsum[t] = 0.0f; lsq[t] = 0.0f; }

    // stage BN1(z1)+ReLU: 16 contiguous halves per thread
    {
        const int r = t >> 2;
        const int k0 = (t & 3) * 16;
        const _Float16* src = z1 + (rowBase + r) * 64 + k0;
        for (int i = 0; i < 16; ++i) {
            const int c = k0 + i;
            float v = (float)src[i] * scale1[c] + shift1[c];
            As[r][c] = (_Float16)(v > 0.0f ? v : 0.0f);
        }
    }
    __syncthreads();

    const int wv = t >> 5, lane = t & 31;
    const int cl = lane & 15;
    const int hs = lane >> 4;
    const int ct = wv;                      // col tile 0..7

    for (int rt = 0; rt < 4; ++rt) {
        const int row0 = rt * 16;
        v8f acc = {};
        for (int ks = 0; ks < 2; ++ks) {
            const int k0 = ks * 32;
            v16h a = load_a_frag(&As[row0 + cl][k0 + (hs << 3)]);
            v16h bf = load_b_frag(&Wt[ct * 16 + cl][k0 + (hs << 3)]);
            acc = __builtin_amdgcn_wmma_f32_16x16x32_f16(
                false, a, false, bf, (short)0, acc, false, false);
        }
        const int o = ct * 16 + cl;
        const float bias = b2[o];
        float s = 0.0f, s2 = 0.0f;
        for (int r8 = 0; r8 < 8; ++r8) {
            const long r = rowBase + row0 + hs * 8 + r8;
            const float z = acc[r8] + bias;
            z2[r * 128 + o] = (_Float16)z;
            s += z; s2 += z * z;
        }
        atomicAdd(&lsum[o], s);
        atomicAdd(&lsq[o], s2);
    }
    __syncthreads();
    if (t < 128) { atomicAdd(&sum2[t], lsum[t]); atomicAdd(&sq2[t], lsq[t]); }
}

// ---------------------------------------------------------------------------
// Kernel 5: inverse-distance weighted pooling + BN2 + ReLU, write (B,128,M)
// transposed output. 2 queries per 256-thread block; thread = (query, out-ch).
// ---------------------------------------------------------------------------
__global__ __launch_bounds__(256) void pool_kernel(
    const _Float16* __restrict__ z2, const float* __restrict__ recW,
    const float* __restrict__ scale2, const float* __restrict__ shift2,
    float* __restrict__ out)
{
    __shared__ float recS[2][KSAMP];
    __shared__ float wsumS[2];
    const int t = threadIdx.x;
    const int q = t >> 7;
    const int o = t & 127;
    const long qid = (long)blockIdx.x * 2 + q;
    const long row0 = qid * KSAMP;

    if (o < KSAMP) recS[q][o] = recW[row0 + o];
    __syncthreads();
    if (o == 0) {
        float s = 0.0f;
        for (int k = 0; k < KSAMP; ++k) s += recS[q][k];
        wsumS[q] = s;
    }
    __syncthreads();

    const float inv = 1.0f / wsumS[q];
    const float sc = scale2[o], sh = shift2[o];
    float acc = 0.0f;
    for (int k = 0; k < KSAMP; ++k) {
        float h = (float)z2[(row0 + k) * 128 + o] * sc + sh;
        h = h > 0.0f ? h : 0.0f;
        acc += recS[q][k] * inv * h;
    }
    const int b = (int)(qid >> 12);
    const int m = (int)(qid & 4095);
    out[(long)(B_BATCH * M_QRY * 3) + ((long)(b * 128 + o)) * M_QRY + m] = acc;
}

// ---------------------------------------------------------------------------
extern "C" void kernel_launch(void* const* d_in, const int* in_sizes, int n_in,
                              void* d_out, int out_size, void* d_ws, size_t ws_size,
                              hipStream_t stream) {
    const float* xyz    = (const float*)d_in[0];
    const float* newxyz = (const float*)d_in[1];
    const float* feat   = (const float*)d_in[2];
    const float* W1     = (const float*)d_in[3];
    const float* b1     = (const float*)d_in[4];
    const float* g1     = (const float*)d_in[5];
    const float* be1    = (const float*)d_in[6];
    const float* W2     = (const float*)d_in[7];
    const float* b2     = (const float*)d_in[8];
    const float* g2     = (const float*)d_in[9];
    const float* be2    = (const float*)d_in[10];
    float* out = (float*)d_out;

    // workspace carve-up (~106 MB)
    int*   idxW  = (int*)d_ws;                      // ROWS
    float* recW  = (float*)(idxW + ROWS);           // ROWS
    float* relW  = recW + ROWS;                     // ROWS*3
    float* stats = relW + (size_t)ROWS * 3;         // 768 floats
    float* sum1 = stats,       *sq1 = stats + 64;
    float* sum2 = stats + 128, *sq2 = stats + 256;
    float* scale1 = stats + 384, *shift1 = stats + 448;
    float* scale2 = stats + 512, *shift2 = stats + 640;
    _Float16* wt1g = (_Float16*)(stats + 768);      // 64*96
    _Float16* wt2g = wt1g + 64 * 96;                // 128*64
    _Float16* z1 = wt2g + 128 * 64;                 // ROWS*64
    _Float16* z2 = z1 + (size_t)ROWS * 64;          // ROWS*128

    const float invCount = 1.0f / (float)ROWS;

    // output[0] = new_xyz passthrough
    (void)hipMemcpyAsync(out, newxyz, (size_t)B_BATCH * M_QRY * 3 * sizeof(float),
                         hipMemcpyDeviceToDevice, stream);

    zero_stats_kernel<<<1, 384, 0, stream>>>(stats);
    prep_weights_kernel<<<1, 256, 0, stream>>>(W1, W2, wt1g, wt2g);
    ballquery_kernel<<<(B_BATCH * M_QRY) / 8, 256, 0, stream>>>(
        xyz, newxyz, idxW, recW, relW);
    layer1_kernel<<<ROWS / 128, 256, 0, stream>>>(
        feat, wt1g, b1, idxW, relW, z1, sum1, sq1);
    finalize_bn_kernel<<<1, 64, 0, stream>>>(
        sum1, sq1, g1, be1, scale1, shift1, 64, invCount);
    layer2_kernel<<<ROWS / 64, 256, 0, stream>>>(
        z1, wt2g, b2, scale1, shift1, z2, sum2, sq2);
    finalize_bn_kernel<<<1, 128, 0, stream>>>(
        sum2, sq2, g2, be2, scale2, shift2, 128, invCount);
    pool_kernel<<<(B_BATCH * M_QRY) / 2, 256, 0, stream>>>(
        z2, recW, scale2, shift2, out);
}
